// MambaAdapter_72172630442418
// MI455X (gfx1250) — matrix operand order repaired and checked by reading, not compiled, not run on hardware
//
#include <hip/hip_runtime.h>
#include <hip/hip_bf16.h>
#include <math.h>

// ---------------------------------------------------------------------------
// Mamba block forward for MI455X (gfx1250, wave32, WMMA).
// GEMMs: v_wmma_f32_16x16x32_f16 + async global->LDS staging (ASYNCcnt).
// ---------------------------------------------------------------------------

typedef __attribute__((ext_vector_type(16))) _Float16 v16h;
typedef __attribute__((ext_vector_type(8)))  _Float16 v8h;
typedef __attribute__((ext_vector_type(4)))  _Float16 v4h;
typedef __attribute__((ext_vector_type(8)))  float    v8f;

union Frag16 { v16h f; v8h h[2]; };

#define D_MODEL 1024
#define D_INNER 2048
#define D_STATE 16
#define D_CONV  4
#define DT_RANK 64
#define BATCH   2
#define SEQLEN  2048
#define ROWS    (BATCH * SEQLEN)     // 4096

// ------------------------------ LayerNorm ----------------------------------
__global__ __launch_bounds__(256)
void ln_to_f16_kernel(const float* __restrict__ x, const float* __restrict__ g,
                      const float* __restrict__ b, _Float16* __restrict__ h16)
{
    const int row = blockIdx.x;
    const float* xr = x + (size_t)row * D_MODEL;
    __shared__ float red[256];

    float s = 0.f;
    for (int i = threadIdx.x; i < D_MODEL; i += 256) s += xr[i];
    red[threadIdx.x] = s; __syncthreads();
    for (int off = 128; off > 0; off >>= 1) {
        if (threadIdx.x < off) red[threadIdx.x] += red[threadIdx.x + off];
        __syncthreads();
    }
    const float mu = red[0] * (1.0f / D_MODEL);
    __syncthreads();

    float v = 0.f;
    for (int i = threadIdx.x; i < D_MODEL; i += 256) {
        float d = xr[i] - mu; v += d * d;
    }
    red[threadIdx.x] = v; __syncthreads();
    for (int off = 128; off > 0; off >>= 1) {
        if (threadIdx.x < off) red[threadIdx.x] += red[threadIdx.x + off];
        __syncthreads();
    }
    const float rstd = rsqrtf(red[0] * (1.0f / D_MODEL) + 1e-5f);

    for (int i = threadIdx.x; i < D_MODEL; i += 256)
        h16[(size_t)row * D_MODEL + i] = (_Float16)(((xr[i] - mu) * rstd) * g[i] + b[i]);
}

// ------------------ weight convert: f32 -> f16, same layout ----------------
__global__ __launch_bounds__(256)
void cvt_f16_kernel(const float* __restrict__ W, _Float16* __restrict__ O,
                    size_t total4)   // total elements / 4
{
    size_t idx = (size_t)blockIdx.x * 256 + threadIdx.x;
    if (idx >= total4) return;
    float4 v = ((const float4*)W)[idx];
    v4h o = { (_Float16)v.x, (_Float16)v.y, (_Float16)v.z, (_Float16)v.w };
    ((v4h*)O)[idx] = o;
}

// --------------------------- WMMA GEMM (f16 -> f32) ------------------------
// C[M,N] = A[M,K] * Bt[N,K]^T.  A row-major [M][K] f16, Bt row-major [N][K].
// Block tile 128x128, BK=64. 8 waves (2x4); each wave: 4x2 tiles of 16x16.
// Tiles staged with GLOBAL_LOAD_ASYNC_TO_LDS_B128 (ASYNCcnt), published to
// the workgroup with s_wait_asynccnt 0 + barrier.
// M,K multiples of 128/64 at all call sites; N may be ragged (N=96 call).
#define BM 128
#define BN 128
#define BK 64

__global__ __launch_bounds__(256)
void gemm_wmma_kernel(const _Float16* __restrict__ A, const _Float16* __restrict__ Bt,
                      float* __restrict__ C, int M, int N, int K)
{
    __shared__ __attribute__((aligned(16))) _Float16 sA[BM][BK];   // 16 KB [m][k]
    __shared__ __attribute__((aligned(16))) _Float16 sB[BN][BK];   // 16 KB [n][k]

    const int tid  = threadIdx.x;
    const int wave = tid >> 5;        // 0..7
    const int lane = tid & 31;
    const int wm   = wave >> 2;       // 0..1  (64-row slabs)
    const int wn   = wave & 3;        // 0..3  (32-col slabs)
    const int half = lane >> 4;       // 0..1
    const int l16  = lane & 15;

    const int block_m = blockIdx.y * BM;
    const int block_n = blockIdx.x * BN;

    const v8f zero8 = {0.f,0.f,0.f,0.f,0.f,0.f,0.f,0.f};
    const v8h zero8h = {(_Float16)0.f,(_Float16)0.f,(_Float16)0.f,(_Float16)0.f,
                        (_Float16)0.f,(_Float16)0.f,(_Float16)0.f,(_Float16)0.f};
    v8f acc[4][2];
    for (int i = 0; i < 4; ++i)
        for (int j = 0; j < 2; ++j) acc[i][j] = zero8;

    v8h* const sBv = (v8h*)&sB[0][0];
    const unsigned ldsA0 = (unsigned)(uintptr_t)&sA[0][0];   // LDS byte offset
    const unsigned ldsB0 = (unsigned)(uintptr_t)&sB[0][0];

    for (int k0 = 0; k0 < K; k0 += BK) {
        // ---- async stage A tile: 1024 x 16B chunks, 4 per thread ----
        #pragma unroll
        for (int t = 0; t < 4; ++t) {
            int i = tid + t * 256;
            int r = i >> 3, c = (i & 7) * 8;           // 8 chunks per 64-wide row
            const _Float16* gp = &A[(size_t)(block_m + r) * K + k0 + c];
            unsigned lo = ldsA0 + (unsigned)i * 16u;
            asm volatile("global_load_async_to_lds_b128 %0, %1, off"
                         :: "v"(lo), "v"(gp) : "memory");
        }
        // ---- async stage B tile (ragged rows -> zero-fill via ds_store) ----
        #pragma unroll
        for (int t = 0; t < 4; ++t) {
            int i = tid + t * 256;
            int r = i >> 3, c = (i & 7) * 8;
            int gn = block_n + r;
            if (gn < N) {
                const _Float16* gp = &Bt[(size_t)gn * K + k0 + c];
                unsigned lo = ldsB0 + (unsigned)i * 16u;
                asm volatile("global_load_async_to_lds_b128 %0, %1, off"
                             :: "v"(lo), "v"(gp) : "memory");
            } else {
                sBv[i] = zero8h;
            }
        }
        asm volatile("s_wait_asynccnt 0x0" ::: "memory");
        __syncthreads();

        #pragma unroll
        for (int ks = 0; ks < BK / 32; ++ks) {
            const int kb = ks * 32;
            // B fragments (32x16): lane -> n = l16; halves give K 0-15/16-31
            Frag16 bfrag[2];
            #pragma unroll
            for (int nt = 0; nt < 2; ++nt) {
                const _Float16* p = &sB[wn * 32 + nt * 16 + l16][kb + half * 16];
                bfrag[nt].h[0] = *(const v8h*)p;
                bfrag[nt].h[1] = *(const v8h*)(p + 8);
            }
            #pragma unroll
            for (int mt = 0; mt < 4; ++mt) {
                // A fragment (16x32): lane -> m = l16; halves give K 0-15/16-31
                const _Float16* p = &sA[wm * 64 + mt * 16 + l16][kb + half * 16];
                Frag16 afrag;
                afrag.h[0] = *(const v8h*)p;
                afrag.h[1] = *(const v8h*)(p + 8);
                #pragma unroll
                for (int nt = 0; nt < 2; ++nt) {
                    acc[mt][nt] = __builtin_amdgcn_wmma_f32_16x16x32_f16(
                        false, afrag.f, false, bfrag[nt].f,
                        (short)0, acc[mt][nt], false, false);
                }
            }
        }
        __syncthreads();
    }

    // store: C/D layout -> VGPR r: row = half*8 + r, col = l16
    #pragma unroll
    for (int mt = 0; mt < 4; ++mt) {
        #pragma unroll
        for (int nt = 0; nt < 2; ++nt) {
            int mb = block_m + wm * 64 + mt * 16;
            int nb = block_n + wn * 32 + nt * 16;
            int col = nb + l16;
            if (col < N) {
                #pragma unroll
                for (int r = 0; r < 8; ++r)
                    C[(size_t)(mb + half * 8 + r) * N + col] = acc[mt][nt][r];
            }
        }
    }
}

// --------------------- causal depthwise conv1d + SiLU ----------------------
__global__ __launch_bounds__(256)
void conv_silu_kernel(const float* __restrict__ xr,      // [ROWS][2*D_INNER]
                      const float* __restrict__ conv_w,  // [D_INNER][1][4]
                      const float* __restrict__ conv_b,
                      float* __restrict__ xh, _Float16* __restrict__ xh16)
{
    size_t gid = (size_t)blockIdx.x * 256 + threadIdx.x;  // over ROWS*D_INNER
    if (gid >= (size_t)ROWS * D_INNER) return;
    int d = (int)(gid % D_INNER);
    int bl = (int)(gid / D_INNER);
    int l = bl % SEQLEN;
    int b = bl / SEQLEN;

    float acc = conv_b[d];
    #pragma unroll
    for (int j = 0; j < D_CONV; ++j) {
        int ls = l - (D_CONV - 1) + j;
        if (ls >= 0)
            acc += conv_w[d * D_CONV + j] *
                   xr[((size_t)(b * SEQLEN + ls)) * (2 * D_INNER) + d];
    }
    float s = acc / (1.f + expf(-acc));                    // SiLU
    xh[gid]   = s;
    xh16[gid] = (_Float16)s;
}

// -------------------------- dt input pack (f16) ----------------------------
__global__ __launch_bounds__(256)
void dt_pack_kernel(const float* __restrict__ xdbl, _Float16* __restrict__ dtin)
{
    size_t idx = (size_t)blockIdx.x * 256 + threadIdx.x;   // over ROWS*DT_RANK
    if (idx >= (size_t)ROWS * DT_RANK) return;
    int j = (int)(idx % DT_RANK);
    int row = (int)(idx / DT_RANK);
    dtin[idx] = (_Float16)xdbl[(size_t)row * (DT_RANK + 2 * D_STATE) + j];
}

// ---------------------- dt = softplus(dt_raw + b_dt) -----------------------
__global__ __launch_bounds__(256)
void dt_softplus_kernel(float* __restrict__ dt, const float* __restrict__ b_dt)
{
    size_t idx = (size_t)blockIdx.x * 256 + threadIdx.x;   // over ROWS*D_INNER
    if (idx >= (size_t)ROWS * D_INNER) return;
    int d = (int)(idx % D_INNER);
    float t = dt[idx] + b_dt[d];
    dt[idx] = (t > 20.f) ? t : log1pf(expf(t));
}

// ---------------------------- selective scan -------------------------------
// 4096 threads, one per (batch, channel). 16 states in registers.
// B/C [L][16] staged through LDS in chunks of 64 timesteps.
#define SCAN_CHUNK 64
__global__ __launch_bounds__(256)
void scan_kernel(const float* __restrict__ xh, const float* __restrict__ dt,
                 const float* __restrict__ xdbl, const float* __restrict__ A_log,
                 float* __restrict__ y)
{
    const int tid = threadIdx.x;
    const int gid = blockIdx.x * 256 + tid;   // 0..4095 ; whole block same batch
    const int d = gid % D_INNER;
    const int b = gid / D_INNER;

    float A[D_STATE];
    #pragma unroll
    for (int n = 0; n < D_STATE; ++n) A[n] = -expf(A_log[(size_t)d * D_STATE + n]);

    float h[D_STATE];
    #pragma unroll
    for (int n = 0; n < D_STATE; ++n) h[n] = 0.f;

    __shared__ float sB[SCAN_CHUNK][D_STATE];
    __shared__ float sC[SCAN_CHUNK][D_STATE];

    for (int l0 = 0; l0 < SEQLEN; l0 += SCAN_CHUNK) {
        __syncthreads();
        for (int i = tid; i < SCAN_CHUNK * D_STATE; i += 256) {
            int ll = i >> 4, n = i & 15;
            size_t row = (size_t)b * SEQLEN + l0 + ll;
            sB[ll][n] = xdbl[row * (DT_RANK + 2 * D_STATE) + DT_RANK + n];
            sC[ll][n] = xdbl[row * (DT_RANK + 2 * D_STATE) + DT_RANK + D_STATE + n];
        }
        __syncthreads();

        for (int li = 0; li < SCAN_CHUNK; ++li) {
            size_t idx = ((size_t)b * SEQLEN + l0 + li) * D_INNER + d;
            float dtv = dt[idx];
            float xv  = xh[idx];
            float yv  = 0.f;
            #pragma unroll
            for (int n = 0; n < D_STATE; ++n) {
                float dA = __expf(dtv * A[n]);
                float dB = dtv * sB[li][n];
                h[n] = dA * h[n] + dB * xv;
                yv  += h[n] * sC[li][n];
            }
            y[idx] = yv;
        }
    }
}

// ------------- gate: y = (y + xh*D) * silu(res), pack to f16 ---------------
__global__ __launch_bounds__(256)
void gate_kernel(const float* __restrict__ y, const float* __restrict__ xh,
                 const float* __restrict__ Dp, const float* __restrict__ xr,
                 _Float16* __restrict__ y16)
{
    size_t idx = (size_t)blockIdx.x * 256 + threadIdx.x;   // over ROWS*D_INNER
    if (idx >= (size_t)ROWS * D_INNER) return;
    int d = (int)(idx % D_INNER);
    size_t row = idx / D_INNER;
    float res = xr[row * (2 * D_INNER) + D_INNER + d];
    float sr  = res / (1.f + expf(-res));
    float v   = (y[idx] + xh[idx] * Dp[d]) * sr;
    y16[idx]  = (_Float16)v;
}

// ----------------------- residual + alpha * out ----------------------------
__global__ __launch_bounds__(256)
void final_kernel(const float* __restrict__ x, const float* __restrict__ outb,
                  const float* __restrict__ alpha, float* __restrict__ out)
{
    size_t idx = (size_t)blockIdx.x * 256 + threadIdx.x;   // over ROWS*D_MODEL
    if (idx >= (size_t)ROWS * D_MODEL) return;
    out[idx] = x[idx] + alpha[0] * outb[idx];
}

// ---------------------------------------------------------------------------
extern "C" void kernel_launch(void* const* d_in, const int* in_sizes, int n_in,
                              void* d_out, int out_size, void* d_ws, size_t ws_size,
                              hipStream_t stream)
{
    const float* x      = (const float*)d_in[0];
    const float* ln_g   = (const float*)d_in[1];
    const float* ln_b   = (const float*)d_in[2];
    const float* W_in   = (const float*)d_in[3];   // [2*D_INNER, D_MODEL]
    const float* W_x    = (const float*)d_in[4];   // [96, D_INNER]
    const float* W_dt   = (const float*)d_in[5];   // [D_INNER, DT_RANK]
    const float* b_dt   = (const float*)d_in[6];
    const float* W_out  = (const float*)d_in[7];   // [D_MODEL, D_INNER]
    const float* conv_w = (const float*)d_in[8];
    const float* conv_b = (const float*)d_in[9];
    const float* A_log  = (const float*)d_in[10];
    const float* D_par  = (const float*)d_in[11];
    const float* alpha  = (const float*)d_in[12];
    float* out = (float*)d_out;

    // -------- workspace carve (all sizes multiples of 256B) --------
    char* w = (char*)d_ws;
    size_t off = 0;
    auto carve = [&](size_t bytes) -> void* {
        void* p = w + off;
        off += (bytes + 255) & ~(size_t)255;
        return p;
    };
    _Float16* h16    = (_Float16*)carve((size_t)ROWS * D_MODEL * 2);        //  8 MB
    _Float16* Win16  = (_Float16*)carve((size_t)D_MODEL * 2 * D_INNER * 2); //  8 MB (contiguous after h16)
    _Float16* Wx16   = (_Float16*)carve((size_t)D_INNER * 96 * 2);
    _Float16* Wdt16  = (_Float16*)carve((size_t)DT_RANK * D_INNER * 2);
    _Float16* Wout16 = (_Float16*)carve((size_t)D_INNER * D_MODEL * 2);     //  4 MB
    float*    xr     = (float*)   carve((size_t)ROWS * 2 * D_INNER * 4);    // 64 MB
    float*    xh     = (float*)   carve((size_t)ROWS * D_INNER * 4);        // 32 MB
    _Float16* xh16   = (_Float16*)carve((size_t)ROWS * D_INNER * 2);        // 16 MB
    float*    xdbl   = (float*)   carve((size_t)ROWS * 96 * 4);
    _Float16* dtin   = (_Float16*)carve((size_t)ROWS * DT_RANK * 2);
    float*    dt     = (float*)   carve((size_t)ROWS * D_INNER * 4);        // 32 MB (softplus in-place)
    float*    y      = (float*)   carve((size_t)ROWS * D_INNER * 4);        // 32 MB
    // reuse dead buffers:
    _Float16* y16    = xh16;            // xh16 dead after GEMM2, the only GEMM that reads it
    float*    outb   = (float*)d_ws;    // aliases h16+Win16 (16 MB, both dead by then)

    // -------- weight conversions to f16 (layout preserved: [N][K]) --------
    {
        size_t t4 = (size_t)(2 * D_INNER) * D_MODEL / 4;
        cvt_f16_kernel<<<(t4 + 255) / 256, 256, 0, stream>>>(W_in, Win16, t4);
    }
    {
        size_t t4 = (size_t)(DT_RANK + 2 * D_STATE) * D_INNER / 4;
        cvt_f16_kernel<<<(t4 + 255) / 256, 256, 0, stream>>>(W_x, Wx16, t4);
    }
    {
        size_t t4 = (size_t)D_INNER * DT_RANK / 4;
        cvt_f16_kernel<<<(t4 + 255) / 256, 256, 0, stream>>>(W_dt, Wdt16, t4);
    }
    {
        size_t t4 = (size_t)D_MODEL * D_INNER / 4;
        cvt_f16_kernel<<<(t4 + 255) / 256, 256, 0, stream>>>(W_out, Wout16, t4);
    }

    // -------- LayerNorm -> h16 --------
    ln_to_f16_kernel<<<ROWS, 256, 0, stream>>>(x, ln_g, ln_b, h16);

    // -------- GEMM1: xr = h16 @ W_in.T  [4096 x 4096] --------
    {
        dim3 grid((2 * D_INNER + BN - 1) / BN, (ROWS + BM - 1) / BM);
        gemm_wmma_kernel<<<grid, 256, 0, stream>>>(h16, Win16, xr,
                                                   ROWS, 2 * D_INNER, D_MODEL);
    }

    // -------- depthwise conv + SiLU --------
    conv_silu_kernel<<<((size_t)ROWS * D_INNER + 255) / 256, 256, 0, stream>>>(
        xr, conv_w, conv_b, xh, xh16);

    // -------- GEMM2: x_dbl = xh16 @ W_x.T  [4096 x 96] --------
    {
        int N = DT_RANK + 2 * D_STATE;
        dim3 grid((N + BN - 1) / BN, (ROWS + BM - 1) / BM);
        gemm_wmma_kernel<<<grid, 256, 0, stream>>>(xh16, Wx16, xdbl,
                                                   ROWS, N, D_INNER);
    }

    // -------- dt path: pack -> GEMM3 -> softplus(+bias) --------
    dt_pack_kernel<<<((size_t)ROWS * DT_RANK + 255) / 256, 256, 0, stream>>>(xdbl, dtin);
    {
        dim3 grid((D_INNER + BN - 1) / BN, (ROWS + BM - 1) / BM);
        gemm_wmma_kernel<<<grid, 256, 0, stream>>>(dtin, Wdt16, dt,
                                                   ROWS, D_INNER, DT_RANK);
    }
    dt_softplus_kernel<<<((size_t)ROWS * D_INNER + 255) / 256, 256, 0, stream>>>(dt, b_dt);

    // -------- selective scan --------
    scan_kernel<<<(BATCH * D_INNER) / 256, 256, 0, stream>>>(xh, dt, xdbl, A_log, y);

    // -------- gate + pack f16 --------
    gate_kernel<<<((size_t)ROWS * D_INNER + 255) / 256, 256, 0, stream>>>(
        y, xh, D_par, xr, y16);

    // -------- GEMM4: outb = y16 @ W_out.T  [4096 x 1024] --------
    {
        dim3 grid((D_MODEL + BN - 1) / BN, (ROWS + BM - 1) / BM);
        gemm_wmma_kernel<<<grid, 256, 0, stream>>>(y16, Wout16, outb,
                                                   ROWS, D_MODEL, D_INNER);
    }

    // -------- residual + alpha * out --------
    final_kernel<<<((size_t)ROWS * D_MODEL + 255) / 256, 256, 0, stream>>>(
        x, outb, alpha, out);

    (void)in_sizes; (void)n_in; (void)out_size; (void)ws_size;
}
